// Int8LinearLayer_67989332296182
// MI455X (gfx1250) — 8-bit matrix operand order repaired and checked by reading, compiled, not run on hardware
//
#include <hip/hip_runtime.h>
#include <stdint.h>

typedef int v8i __attribute__((ext_vector_type(8)));
typedef int v4i __attribute__((ext_vector_type(4)));

#define BM 128
#define BN 128
#define BK 64
#define LDST 80
#define TILEBYTES (128 * LDST)

#if __has_builtin(__builtin_amdgcn_global_load_async_to_lds_b128)
  #define USE_ASYNC_LDS 1
#else
  #define USE_ASYNC_LDS 0
#endif

typedef __attribute__((address_space(1))) v4i global_v4i;
typedef __attribute__((address_space(3))) v4i shared_v4i;

__device__ __forceinline__ void wait_async_lds() {
#if __has_builtin(__builtin_amdgcn_s_wait_asynccnt)
    __builtin_amdgcn_s_wait_asynccnt(0);
#else
    asm volatile("s_wait_asynccnt 0" ::: "memory");
#endif
}

__global__ __launch_bounds__(256)
void Int8Linear_absmax_kernel(const float* __restrict__ x,
                              unsigned int* __restrict__ absbits, int n4) {
    int tid = blockIdx.x * blockDim.x + threadIdx.x;
    int stride = gridDim.x * blockDim.x;
    const float4* x4 = (const float4*)x;
    float m = 0.0f;
    for (int i = tid; i < n4; i += stride) {
        float4 v = x4[i];
        m = fmaxf(m, fmaxf(fmaxf(fabsf(v.x), fabsf(v.y)),
                           fmaxf(fabsf(v.z), fabsf(v.w))));
    }
    #pragma unroll
    for (int off = 16; off > 0; off >>= 1)
        m = fmaxf(m, __shfl_xor(m, off, 32));
    if ((threadIdx.x & 31) == 0)
        atomicMax(absbits, __float_as_uint(m));
}

__global__ __launch_bounds__(256)
void Int8Linear_quant_kernel(const float* __restrict__ x,
                             const unsigned int* __restrict__ absbits,
                             int* __restrict__ xq4, int n4) {
    float xs = fmaxf(__uint_as_float(*absbits) / 127.0f, 1e-5f);
    float inv = 1.0f / xs;
    int tid = blockIdx.x * blockDim.x + threadIdx.x;
    int stride = gridDim.x * blockDim.x;
    const float4* x4 = (const float4*)x;
    for (int i = tid; i < n4; i += stride) {
        float4 v = x4[i];
        int q0 = min(127, max(-128, __float2int_rn(v.x * inv)));
        int q1 = min(127, max(-128, __float2int_rn(v.y * inv)));
        int q2 = min(127, max(-128, __float2int_rn(v.z * inv)));
        int q3 = min(127, max(-128, __float2int_rn(v.w * inv)));
        xq4[i] = (q0 & 0xFF) | ((q1 & 0xFF) << 8) | ((q2 & 0xFF) << 16) | (q3 << 24);
    }
}

__global__ __launch_bounds__(256)
void Int8Linear_gemm_kernel(const int8_t* __restrict__ xq,
                            const int8_t* __restrict__ w,
                            const float* __restrict__ wscale,
                            const float* __restrict__ bias,
                            const unsigned int* __restrict__ absbits,
                            float* __restrict__ out,
                            int M, int N, int K) {
    __shared__ __align__(16) char lds[2][2 * TILEBYTES];

    const int tid   = threadIdx.x;
    const int lane  = tid & 31;
    const int wid   = tid >> 5;
    const int waveM = wid >> 2;
    const int waveN = wid & 3;
    const int m0 = blockIdx.y * BM;
    const int n0 = blockIdx.x * BN;

    const float xs = fmaxf(__uint_as_float(*absbits) / 127.0f, 1e-5f);

    v8i acc[4][2];
    #pragma unroll
    for (int mi = 0; mi < 4; ++mi)
        #pragma unroll
        for (int ni = 0; ni < 2; ++ni)
            #pragma unroll
            for (int j = 0; j < 8; ++j)
                acc[mi][ni][j] = 0;

    const int kSteps = K / BK;

    const int chunk0 = tid * 2;
    const int row0   = chunk0 >> 2;
    const int co0    = (chunk0 & 3) * 16;
    const int row1   = (chunk0 + 1) >> 2;
    const int co1    = ((chunk0 + 1) & 3) * 16;

    auto loadTiles = [&](int kb, int stage) {
        const int8_t* ga0 = xq + (size_t)(m0 + row0) * K + (size_t)kb * BK + co0;
        const int8_t* ga1 = xq + (size_t)(m0 + row1) * K + (size_t)kb * BK + co1;
        const int8_t* gb0 = w  + (size_t)(n0 + row0) * K + (size_t)kb * BK + co0;
        const int8_t* gb1 = w  + (size_t)(n0 + row1) * K + (size_t)kb * BK + co1;
        char* la0 = &lds[stage][row0 * LDST + co0];
        char* la1 = &lds[stage][row1 * LDST + co1];
        char* lb0 = &lds[stage][TILEBYTES + row0 * LDST + co0];
        char* lb1 = &lds[stage][TILEBYTES + row1 * LDST + co1];
#if USE_ASYNC_LDS
        __builtin_amdgcn_global_load_async_to_lds_b128((global_v4i*)ga0, (shared_v4i*)la0, 0, 0);
        __builtin_amdgcn_global_load_async_to_lds_b128((global_v4i*)ga1, (shared_v4i*)la1, 0, 0);
        __builtin_amdgcn_global_load_async_to_lds_b128((global_v4i*)gb0, (shared_v4i*)lb0, 0, 0);
        __builtin_amdgcn_global_load_async_to_lds_b128((global_v4i*)gb1, (shared_v4i*)lb1, 0, 0);
#else
        int4 va0 = *(const int4*)ga0;
        int4 va1 = *(const int4*)ga1;
        int4 vb0 = *(const int4*)gb0;
        int4 vb1 = *(const int4*)gb1;
        *(int4*)la0 = va0;
        *(int4*)la1 = va1;
        *(int4*)lb0 = vb0;
        *(int4*)lb1 = vb1;
#endif
    };

    auto loadFragA = [&](const char* tile, int rowBase) -> v8i {
        int r  = rowBase + (lane & 15);
        int kl = (lane >> 4) * 8;
        v8i f;
        #pragma unroll
        for (int p = 0; p < 4; ++p) {
            uint2 d = *(const uint2*)(tile + r * LDST + p * 16 + kl);
            f[2 * p]     = (int)d.x;
            f[2 * p + 1] = (int)d.y;
        }
        return f;
    };

    auto loadFragB = [&](const char* tile, int rowBase) -> v8i {
        int r  = rowBase + (lane & 15);
        int kl = (lane >> 4) * 8;
        v8i f;
        #pragma unroll
        for (int p = 0; p < 4; ++p) {
            uint2 d = *(const uint2*)(tile + r * LDST + p * 16 + kl);
            f[2 * p]     = (int)d.x;
            f[2 * p + 1] = (int)d.y;
        }
        return f;
    };

    auto computeStage = [&](int stage) {
        const char* aT = &lds[stage][0];
        const char* bT = &lds[stage][TILEBYTES];
        v8i af[4];
        #pragma unroll
        for (int mi = 0; mi < 4; ++mi)
            af[mi] = loadFragA(aT, waveM * 64 + mi * 16);
        v8i bf[2];
        #pragma unroll
        for (int ni = 0; ni < 2; ++ni)
            bf[ni] = loadFragB(bT, waveN * 32 + ni * 16);
        #pragma unroll
        for (int mi = 0; mi < 4; ++mi)
            #pragma unroll
            for (int ni = 0; ni < 2; ++ni)
                acc[mi][ni] = __builtin_amdgcn_wmma_i32_16x16x64_iu8(
                     true, af[mi],  true, bf[ni],
                    acc[mi][ni],  false,  false);
    };

    loadTiles(0, 0);
#if USE_ASYNC_LDS
    wait_async_lds();
#endif
    __syncthreads();
    for (int kb = 0; kb < kSteps; ++kb) {
        int cur = kb & 1;
        if (kb + 1 < kSteps)
            loadTiles(kb + 1, cur ^ 1);
        computeStage(cur);
#if USE_ASYNC_LDS
        wait_async_lds();
#endif
        __syncthreads();
    }

    #pragma unroll
    for (int mi = 0; mi < 4; ++mi) {
        #pragma unroll
        for (int ni = 0; ni < 2; ++ni) {
            int col = n0 + waveN * 32 + ni * 16 + (lane & 15);
            float sc = xs * wscale[col];
            float bs = bias[col];
            int rbase = m0 + waveM * 64 + mi * 16 + ((lane >> 4) * 8);
            #pragma unroll
            for (int v = 0; v < 8; ++v) {
                int row = rbase + v;
                out[(size_t)row * N + col] = (float)acc[mi][ni][v] * sc + bs;
            }
        }
    }
}

extern "C" void kernel_launch(void* const* d_in, const int* in_sizes, int n_in,
                              void* d_out, int out_size, void* d_ws, size_t ws_size,
                              hipStream_t stream) {
    const float*   x      = (const float*)d_in[0];
    const int8_t*  w      = (const int8_t*)d_in[1];
    const float*   wscale = (const float*)d_in[2];
    const float*   bias   = (const float*)d_in[3];
    float*         out    = (float*)d_out;

    const int xcount = in_sizes[0];
    const int OUT    = in_sizes[2];
    const int K      = in_sizes[1] / OUT;
    const int M      = xcount / K;

    unsigned int* absbits = (unsigned int*)d_ws;
    int8_t*       xq      = (int8_t*)d_ws + 256;

    (void)hipMemsetAsync(d_ws, 0, 256, stream);

    const int n4 = xcount / 4;
    Int8Linear_absmax_kernel<<<1024, 256, 0, stream>>>(x, absbits, n4);
    Int8Linear_quant_kernel<<<2048, 256, 0, stream>>>(x, absbits, (int*)xq, n4);

    dim3 grid(OUT / BN, M / BM);
    Int8Linear_gemm_kernel<<<grid, 256, 0, stream>>>(xq, w, wscale, bias,
                                                     absbits, out, M, OUT, K);
}
